// HungarianMatcher_16338055594381
// MI455X (gfx1250) — matrix operand order repaired and checked by reading, not compiled, not run on hardware
//
#include <hip/hip_runtime.h>
#include <hip/hip_bf16.h>
#include <stdint.h>

// ---------------------------------------------------------------------------
// HungarianMatcher for MI455X (gfx1250), wave32 / WMMA / async-LDS.
// Phase A : softmax(logits/0.7) -> f16 prob matrix (2400 x 96, zero-padded K).
// Phase A': one-hot(labels) -> f16 matrix (800 x 96), built once, L2-resident.
// Phase B : fused cost tiles: class cost = prob x one-hot via
//           v_wmma_f32_16x16x32_f16 (3 K-steps, B fragments loaded b128);
//           point cost = L1 on coord tiles DMA'd into LDS with
//           global_load_async_to_lds_b128, float4-vectorized epilogue.
// Phase C : 8 parallel Jonker-Volgenant solves, one workgroup per batch,
//           block-wide argmin reductions, L2-resident cost rows + prefetch.
// ---------------------------------------------------------------------------

typedef __attribute__((ext_vector_type(16))) _Float16 v16h;
typedef __attribute__((ext_vector_type(8)))  float    v8f;

constexpr int B_  = 8;
constexpr int Q_  = 300;
constexpr int K_  = 81;
constexpr int KP  = 96;     // K padded to 3 * 32 for WMMA K-steps
constexpr int T_  = 100;
constexpr int P2  = 32;     // 16 points * 2 coords
constexpr int BQ  = B_ * Q_;   // 2400
constexpr int BT  = B_ * T_;   // 800
constexpr float TEMP_INV = 1.0f / 0.7f;
constexpr float W_PT     = 10.0f;   // max(1, 10 * 0.95^0), EPOCH = 0
constexpr float INF_     = 1e18f;

// ------------------------------ Phase A ------------------------------------
__global__ void __launch_bounds__(128)
softmax_f16_kernel(const float* __restrict__ logits, _Float16* __restrict__ prob) {
    const int row = blockIdx.x;          // 0..2399
    const int t   = threadIdx.x;         // 0..127
    __shared__ float red[128];

    float x = -3.0e38f;
    if (t < K_) x = logits[row * K_ + t] * TEMP_INV;
    red[t] = x;
    __syncthreads();
    for (int off = 64; off > 0; off >>= 1) {
        if (t < off) red[t] = fmaxf(red[t], red[t + off]);
        __syncthreads();
    }
    const float mx = red[0];
    __syncthreads();

    const float e = (t < K_) ? __expf(x - mx) : 0.0f;
    red[t] = e;
    __syncthreads();
    for (int off = 64; off > 0; off >>= 1) {
        if (t < off) red[t] += red[t + off];
        __syncthreads();
    }
    const float inv = 1.0f / red[0];

    if (t < KP) prob[row * KP + t] = (_Float16)((t < K_) ? e * inv : 0.0f);
}

// ------------------------------ Phase A' -----------------------------------
// one-hot^T (BT x KP f16): oh[n*KP + k] = (labels[n] == k).  150 KB, built
// once, then reused by all 150 row-tiles per column tile out of L2.
__global__ void __launch_bounds__(128)
onehot_kernel(const int* __restrict__ labels, _Float16* __restrict__ oh) {
    const int n = blockIdx.x;            // 0..799
    const int k = threadIdx.x;           // 0..127
    if (k < KP) oh[n * KP + k] = (labels[n] == k) ? (_Float16)1.0f : (_Float16)0.0f;
}

// ------------------------------ Phase B ------------------------------------
// One wave per 16x16 output tile.  grid = (BT/16, BQ/16), block = 32.
__global__ void __launch_bounds__(32)
cost_tile_kernel(const _Float16* __restrict__ prob,   // BQ x KP (f16)
                 const _Float16* __restrict__ oh,     // BT x KP (f16)
                 const float*    __restrict__ qpts,   // BQ x 32 (f32)
                 const float*    __restrict__ tpts,   // BT x 32 (f32)
                 float*          __restrict__ Cout)   // BQ x BT
{
    __shared__ float sA[16 * P2];   // 2 KB: query coord tile
    __shared__ float sB[16 * P2];   // 2 KB: target coord tile

    const int lane = threadIdx.x;                 // 0..31
    const int tn   = blockIdx.x, tm = blockIdx.y;
    const int m0   = tm * 16,    n0 = tn * 16;
    const int half = (lane < 16) ? 0 : 1;

    // ---- kick off async DMA of both coordinate tiles into LDS -------------
    // Each instruction moves 32 lanes x 16 B = 512 B; 4 per 2 KB tile.
    const unsigned ldsA = (unsigned)(uintptr_t)(&sA[0]);
    const unsigned ldsB = (unsigned)(uintptr_t)(&sB[0]);
    const unsigned long long gA =
        (unsigned long long)(uintptr_t)(qpts + (size_t)m0 * P2);
    const unsigned long long gB =
        (unsigned long long)(uintptr_t)(tpts + (size_t)n0 * P2);
    #pragma unroll
    for (int i = 0; i < 4; ++i) {
        const unsigned off = (unsigned)(lane * 16 + i * 512);
        asm volatile("global_load_async_to_lds_b128 %0, %1, %2"
                     :: "v"(ldsA + off), "v"(off), "s"(gA) : "memory");
        asm volatile("global_load_async_to_lds_b128 %0, %1, %2"
                     :: "v"(ldsB + off), "v"(off), "s"(gB) : "memory");
    }

    // ---- class cost = prob x one-hot, 3 WMMA K-steps ----------------------
    const int mrow = m0 + (lane & 15);
    const int ncol = n0 + (lane & 15);

    v8f acc = {};
    #pragma unroll
    for (int s = 0; s < 3; ++s) {
        // A fragment: lanes 0-15 hold K = {s32+0..7, s32+16..23} of row m,
        // lanes 16-31 hold K = {s32+8..15, s32+24..31} (16-bit A layout).
        union { uint4 u[2]; v16h v; } af;
        const _Float16* rp = prob + (size_t)mrow * KP + s * 32 + half * 8;
        af.u[0] = *(const uint4*)(rp);
        af.u[1] = *(const uint4*)(rp + 16);

        // B fragment: lanes 0-15 hold K = {s32+0..7, s32+16..23} of column n,
        // lanes 16-31 hold K = {s32+8..15, s32+24..31} (same order as A).
        union { uint4 u[2]; v16h v; } bf;
        const _Float16* bp = oh + (size_t)ncol * KP + s * 32 + half * 8;
        bf.u[0] = *(const uint4*)(bp);
        bf.u[1] = *(const uint4*)(bp + 16);

        acc = __builtin_amdgcn_wmma_f32_16x16x32_f16(
                  false, af.v, false, bf.v, (short)0, acc, false, false);
    }

    // ---- wait for the coordinate DMA, then fuse L1 point cost -------------
    asm volatile("s_wait_asynccnt 0x0" ::: "memory");

    const int N = lane & 15;
    float4 bcol[8];                    // target point column lives in VGPRs
    #pragma unroll
    for (int d4 = 0; d4 < 8; ++d4)
        bcol[d4] = *(const float4*)(&sB[N * P2 + d4 * 4]);

    #pragma unroll
    for (int j = 0; j < 8; ++j) {
        const int M = j + 8 * half;               // WMMA C/D layout
        float l1 = 0.0f;
        #pragma unroll
        for (int d4 = 0; d4 < 8; ++d4) {
            const float4 a = *(const float4*)(&sA[M * P2 + d4 * 4]);
            l1 += fabsf(a.x - bcol[d4].x) + fabsf(a.y - bcol[d4].y)
                + fabsf(a.z - bcol[d4].z) + fabsf(a.w - bcol[d4].w);
        }
        // COST_CLASS_W * (-prob) + w_pt * L1
        Cout[(size_t)(m0 + M) * BT + (n0 + N)] = -acc[j] + W_PT * l1;
    }
}

// ------------------------------ Phase C ------------------------------------
// One workgroup per batch; Jonker-Volgenant with block-parallel column scan.
constexpr int S_   = 300;   // square problem size = max(Q, T)
constexpr int BDIM = 512;

__global__ void __launch_bounds__(BDIM)
hungarian_kernel(const float* __restrict__ Cfull,
                 float* __restrict__ outRow, float* __restrict__ outCol)
{
    const int b   = blockIdx.x;
    const int tid = threadIdx.x;
    const float* Cb = Cfull + (size_t)b * Q_ * BT + b * T_;  // Cb[r*BT + t]

    __shared__ float u[S_ + 1], v[S_ + 1], minv[S_ + 1];
    __shared__ int   p[S_ + 1], way[S_ + 1];
    __shared__ unsigned char used[S_ + 1];
    __shared__ float redV[BDIM];
    __shared__ int   redI[BDIM];
    __shared__ int   col4row[S_];
    __shared__ float fillv;
    __shared__ int   sI0, sJ0;

    // pad value = max(cost) + 1
    float mx = -3.0e38f;
    for (int idx = tid; idx < Q_ * T_; idx += BDIM) {
        const int r = idx / T_, t = idx - r * T_;
        mx = fmaxf(mx, Cb[(size_t)r * BT + t]);
    }
    redV[tid] = mx;
    __syncthreads();
    for (int off = 256; off > 0; off >>= 1) {
        if (tid < off) redV[tid] = fmaxf(redV[tid], redV[tid + off]);
        __syncthreads();
    }
    if (tid == 0) fillv = redV[0] + 1.0f;

    for (int jj = tid; jj <= S_; jj += BDIM) { u[jj] = 0.f; v[jj] = 0.f; p[jj] = 0; way[jj] = 0; }
    __syncthreads();

    for (int i = 1; i <= S_; ++i) {
        if (tid == 0) { p[0] = i; sJ0 = 0; }
        for (int jj = tid; jj <= S_; jj += BDIM) { minv[jj] = INF_; used[jj] = 0; }
        // row i-1 is about to be scanned repeatedly: pull it into L2/L0
        if (tid < T_) __builtin_prefetch(&Cb[(size_t)(i - 1) * BT + tid], 0, 1);
        __syncthreads();

        while (true) {
            if (tid == 0) { used[sJ0] = 1; sI0 = p[sJ0]; }
            __syncthreads();
            const int   i0  = sI0;
            const int   j0  = sJ0;
            const float ui0 = u[i0];

            float cand = INF_; int candj = 0;
            if (tid < S_) {
                const int j = tid + 1;
                if (!used[j]) {
                    const float cu =
                        (((j - 1) < T_) ? Cb[(size_t)(i0 - 1) * BT + (j - 1)] : fillv)
                        - ui0 - v[j];
                    if (cu < minv[j]) { minv[j] = cu; way[j] = j0; }
                    cand = minv[j]; candj = j;
                }
            }
            redV[tid] = cand; redI[tid] = candj;
            __syncthreads();
            for (int off = 256; off > 0; off >>= 1) {
                if (tid < off && redV[tid + off] < redV[tid]) {
                    redV[tid] = redV[tid + off]; redI[tid] = redI[tid + off];
                }
                __syncthreads();
            }
            const float delta = redV[0];
            const int   j1    = redI[0];

            if (tid <= S_) {
                if (used[tid]) { u[p[tid]] += delta; v[tid] -= delta; }
                else if (tid >= 1) minv[tid] -= delta;
            }
            if (tid == 0) sJ0 = j1;
            __syncthreads();
            if (p[j1] == 0) break;     // uniform: p is stable inside the loop
        }

        if (tid == 0) {                // augment along `way`
            int j0 = sJ0;
            while (j0) { const int jp = way[j0]; p[j0] = p[jp]; j0 = jp; }
        }
        __syncthreads();
    }

    for (int j = tid + 1; j <= S_; j += BDIM) col4row[p[j] - 1] = j - 1;
    __syncthreads();
    if (tid == 0) {                    // compact: keep cols < T (exactly T kept)
        int k = 0;
        for (int r = 0; r < S_; ++r) {
            const int c = col4row[r];
            if (c < T_) { outRow[b * T_ + k] = (float)r; outCol[b * T_ + k] = (float)c; ++k; }
        }
    }
}

// ------------------------------ launcher -----------------------------------
extern "C" void kernel_launch(void* const* d_in, const int* in_sizes, int n_in,
                              void* d_out, int out_size, void* d_ws, size_t ws_size,
                              hipStream_t stream) {
    (void)in_sizes; (void)n_in; (void)out_size; (void)ws_size;

    const float* pred_logits = (const float*)d_in[0];   // (8,300,81)  f32
    const float* pred_coords = (const float*)d_in[1];   // (8,300,16,2) f32
    const int*   tgt_labels  = (const int*)  d_in[2];   // (8,100)      int
    const float* tgt_coords  = (const float*)d_in[3];   // (8,100,16,2) f32

    float* out    = (float*)d_out;
    float* outRow = out;                 // (8,100)
    float* outCol = out + BT;            // (8,100)
    float* Cout   = out + 2 * BT;        // (8,300,800)

    _Float16* prob = (_Float16*)d_ws;            // 2400 x 96 f16 = 460.8 KB
    _Float16* oh   = prob + (size_t)BQ * KP;     //  800 x 96 f16 = 153.6 KB

    hipLaunchKernelGGL(softmax_f16_kernel, dim3(BQ), dim3(128), 0, stream,
                       pred_logits, prob);
    hipLaunchKernelGGL(onehot_kernel, dim3(BT), dim3(128), 0, stream,
                       tgt_labels, oh);
    hipLaunchKernelGGL(cost_tile_kernel, dim3(BT / 16, BQ / 16), dim3(32), 0, stream,
                       prob, oh, pred_coords, tgt_coords, Cout);
    hipLaunchKernelGGL(hungarian_kernel, dim3(B_), dim3(BDIM), 0, stream,
                       Cout, outRow, outCol);
}
